// Hyperpep_12695923327693
// MI455X (gfx1250) — matrix operand order repaired and checked by reading, not compiled, not run on hardware
//
#include <hip/hip_runtime.h>

// ---------------------------------------------------------------------------
// Hyperpep forward on gfx1250 (MI455X), wave32 + WMMA bf16.
// Pipeline: 2x hypergraph layers -> residue ARMA chain conv -> LN -> pool -> MLP
// All dense GEMMs use v_wmma_f32_16x16x32_bf16 (fp32 accum). Scatter-means are
// float atomics; mean division is folded into the GEMM A-load (row scale).
// Chain propagation is a gather (<=2 neighbors), no atomics.
// W is staged in LDS column-major (fragment-ready) so each lane's B fragment
// is one contiguous 32-byte run -> 2x ds_load_b128, no b16 repacking.
// ---------------------------------------------------------------------------

#define HID   128
#define KMAX  160            // max padded K (148 -> 160)
#define KST   168            // per-column LDS pitch: 336B = 21*16 (b128-aligned)
#define SA_ST 40             // LDS stride for A chunk (32 + pad)

typedef __attribute__((ext_vector_type(16))) __bf16 v16bf;
typedef __attribute__((ext_vector_type(8)))  float  v8f;

__device__ __forceinline__ unsigned short f2bf(float f) {
  union { float f; unsigned u; } c; c.f = f;
  unsigned u = c.u;
  return (unsigned short)((u + 0x7FFFu + ((u >> 16) & 1u)) >> 16); // RNE
}

// ---------------------------------------------------------------------------
// C[M,128] = act( [A1 | A2] @ W + bias + addend ), optional 1/max(cnt,1) scale
// on A1 rows (folds scatter_mean). Block = 256 threads (8 waves), 64 rows.
// Wave w owns cols [16w,16w+16), 4 row-tiles of 16 -> 4 WMMA per K-chunk.
// ---------------------------------------------------------------------------
__launch_bounds__(256)
__global__ void gemm_bias_act(
    const float* __restrict__ A1, int K1,
    const float* __restrict__ A2, int K2,          // A2 may be null (K2==0)
    const float* __restrict__ rowCnt,              // null or per-row count
    const float* __restrict__ W,                   // [K,128] row-major in global
    const float* __restrict__ bias,                // null or [128]
    const float* __restrict__ addend,              // null or [M,128]
    float* __restrict__ C, int Mrows, int relu)
{
  __shared__ unsigned short sW[HID * KST];   // column-major: sW[n*KST + k]
  __shared__ unsigned short sA[64 * SA_ST];

  const int K  = K1 + K2;
  const int Kp = (K + 31) & ~31;
  const int tid  = threadIdx.x;
  const int lane = tid & 31;
  const int wave = tid >> 5;                 // 0..7
  const int rowBase = blockIdx.x * 64;
  const int c0 = wave * 16;

  // Stage full W (bf16, column-major) into LDS once.
  for (int idx = tid; idx < Kp * HID; idx += 256) {
    int k = idx >> 7, n = idx & 127;
    float v = (k < K) ? W[k * HID + n] : 0.f;
    sW[n * KST + k] = f2bf(v);
  }

  v8f acc[4];
  #pragma unroll
  for (int t = 0; t < 4; ++t) acc[t] = v8f{};

  for (int kc = 0; kc < Kp; kc += 32) {
    __syncthreads();
    // Stage A chunk: 64 rows x 32 k (concat + row-scale + zero pad).
    for (int idx = tid; idx < 64 * 32; idx += 256) {
      int rl = idx >> 5, kk = idx & 31;
      int row = rowBase + rl, k = kc + kk;
      float v = 0.f;
      if (row < Mrows && k < K) {
        if (k < K1) {
          v = A1[(long)row * K1 + k];
          if (rowCnt) v *= 1.f / fmaxf(rowCnt[row], 1.f);
        } else {
          v = A2[(long)row * K2 + (k - K1)];
        }
      }
      sA[rl * SA_ST + kk] = f2bf(v);
      // Prefetch next K-chunk of the A stream (HBM-resident) while computing.
      int kn = k + 32;
      if (row < Mrows && kn < K) {
        if (kn < K1)      __builtin_prefetch(&A1[(long)row * K1 + kn], 0, 0);
        else              __builtin_prefetch(&A2[(long)row * K2 + (kn - K1)], 0, 0);
      }
    }
    __syncthreads();

    // B fragment (32x16 bf16): lane = column n = c0+(lane&15);
    // lanes 0-15 hold K kc..kc+15, lanes 16-31 hold K kc+16..kc+31;
    // VGPR v = pair {2v,2v+1}. Contiguous in LDS -> 2x ds_load_b128.
    union { v16bf v; unsigned short u[16]; } bf;
    {
      int n  = c0 + (lane & 15);
      int kb = kc + (lane >> 4) * 16;
      const unsigned short* p = &sW[n * KST + kb];
      #pragma unroll
      for (int vi = 0; vi < 16; ++vi) bf.u[vi] = p[vi];
    }
    // A fragment per row tile (16x32 bf16, contiguous K per lane half, matches B) + WMMA.
    #pragma unroll
    for (int rt = 0; rt < 4; ++rt) {
      union { v16bf v; unsigned short u[16]; } af;
      int m  = rt * 16 + (lane & 15);
      int hi = lane >> 4;
      const unsigned short* pa = &sA[m * SA_ST + hi * 16];
      #pragma unroll
      for (int vi = 0; vi < 8; ++vi) af.u[vi] = pa[vi];          // K kb..kb+7
      #pragma unroll
      for (int vi = 0; vi < 8; ++vi) af.u[8 + vi] = pa[8 + vi]; // K kb+8..+15
      acc[rt] = __builtin_amdgcn_wmma_f32_16x16x32_bf16(
          false, af.v, false, bf.v, (short)0, acc[rt], false, false);
    }
  }

  // Epilogue: C layout — VGPR j: lanes0-15 row j, lanes16-31 row j+8; n=lane&15.
  int n = c0 + (lane & 15);
  float bv = bias ? bias[n] : 0.f;
  #pragma unroll
  for (int rt = 0; rt < 4; ++rt) {
    #pragma unroll
    for (int j = 0; j < 8; ++j) {
      int row = rowBase + rt * 16 + (lane >> 4) * 8 + j;
      if (row < Mrows) {
        float v = acc[rt][j] + bv;
        if (addend) v += addend[(long)row * HID + n];
        if (relu) v = fmaxf(v, 0.f);
        C[(long)row * HID + n] = v;
      }
    }
  }
}

// ---------------------------------------------------------------------------
__global__ void fill_zero_f(float* p, long n) {
  long i = (long)blockIdx.x * blockDim.x + threadIdx.x;
  long s = (long)gridDim.x * blockDim.x;
  for (; i < n; i += s) p[i] = 0.f;
}

__global__ void count_deg(const int* __restrict__ fg, const int* __restrict__ rs,
                          float* cnt_h, float* cnt_e, int M) {
  int j = blockIdx.x * 256 + threadIdx.x;
  if (j < M) {
    atomicAdd(&cnt_e[rs[j]], 1.f);
    atomicAdd(&cnt_h[fg[j]], 1.f);
  }
}

// dst[sidx[j]] += src[gidx[j]]  (row dim D, multiple of 4)
__global__ void scatter_add_rows(const float* __restrict__ src,
                                 const int* __restrict__ gidx,
                                 const int* __restrict__ sidx,
                                 float* __restrict__ dst, int M, int D) {
  long t = (long)blockIdx.x * 256 + threadIdx.x;
  int chunks = D >> 2;
  long total = (long)M * chunks;
  if (t >= total) return;
  int j = (int)(t / chunks);
  int d = (int)(t % chunks) << 2;
  float4 v = *(const float4*)(src + (long)gidx[j] * D + d);
  float* p = dst + (long)sidx[j] * D + d;
  atomicAdd(p + 0, v.x); atomicAdd(p + 1, v.y);
  atomicAdd(p + 2, v.z); atomicAdd(p + 3, v.w);
}

__global__ void layernorm_rows(float* __restrict__ x, const float* __restrict__ g,
                               const float* __restrict__ b, int rows) {
  __shared__ float sd[128];
  int row = blockIdx.x;
  if (row >= rows) return;
  int f = threadIdx.x;
  float v = x[(long)row * HID + f];
  sd[f] = v; __syncthreads();
  for (int s = 64; s > 0; s >>= 1) { if (f < s) sd[f] += sd[f + s]; __syncthreads(); }
  float mean = sd[0] * (1.f / 128.f);
  __syncthreads();
  float d = v - mean;
  sd[f] = d * d; __syncthreads();
  for (int s = 64; s > 0; s >>= 1) { if (f < s) sd[f] += sd[f + s]; __syncthreads(); }
  float var = sd[0] * (1.f / 128.f);
  x[(long)row * HID + f] = d * rsqrtf(var + 1e-5f) * g[f] + b[f];
}

__global__ void build_offsets(const int* __restrict__ nh, int* off, int B) {
  if (blockIdx.x == 0 && threadIdx.x == 0) {
    int a = 0; off[0] = 0;
    for (int i = 0; i < B; ++i) { a += nh[i]; off[i + 1] = a; }
  }
}

__global__ void build_seg(const int* __restrict__ off, int* seg) {
  int b = blockIdx.x;
  for (int i = off[b] + threadIdx.x; i < off[b + 1]; i += blockDim.x) seg[i] = b;
}

// out[i] = relu( sum_nbr norm(i,j)*xin[j] + add[i] )  — chain graph gather.
__global__ void prop_combine(const float* __restrict__ xin,
                             const float* __restrict__ add,
                             const int* __restrict__ seg,
                             const int* __restrict__ off,
                             float* __restrict__ out, int Erows) {
  long t = (long)blockIdx.x * 256 + threadIdx.x;
  long total = (long)Erows * 32;
  if (t >= total) return;
  int i = (int)(t >> 5);
  int d = ((int)t & 31) << 2;
  int b = seg[i];
  int lo = off[b], hi = off[b + 1];
  int hp = (i > lo), hn = (i + 1 < hi);
  int deg = hp + hn;
  float di = deg > 0 ? rsqrtf((float)deg) : 0.f;
  float4 s = {0.f, 0.f, 0.f, 0.f};
  if (hp) {
    int j = i - 1; int dj = 1 + (j > lo);
    float nrm = di * rsqrtf((float)dj);
    float4 xv = *(const float4*)(xin + (long)j * HID + d);
    s.x += nrm * xv.x; s.y += nrm * xv.y; s.z += nrm * xv.z; s.w += nrm * xv.w;
  }
  if (hn) {
    int j = i + 1; int dj = 1 + (j + 1 < hi);
    float nrm = di * rsqrtf((float)dj);
    float4 xv = *(const float4*)(xin + (long)j * HID + d);
    s.x += nrm * xv.x; s.y += nrm * xv.y; s.z += nrm * xv.z; s.w += nrm * xv.w;
  }
  float4 a = *(const float4*)(add + (long)i * HID + d);
  float4 o;
  o.x = fmaxf(s.x + a.x, 0.f); o.y = fmaxf(s.y + a.y, 0.f);
  o.z = fmaxf(s.z + a.z, 0.f); o.w = fmaxf(s.w + a.w, 0.f);
  *(float4*)(out + (long)i * HID + d) = o;
}

__global__ void pool_sum(const float* __restrict__ r, const int* __restrict__ off,
                         float* __restrict__ g) {
  int b = blockIdx.x, f = threadIdx.x;
  float s = 0.f;
  for (int i = off[b]; i < off[b + 1]; ++i) s += r[(long)i * HID + f];
  g[b * HID + f] = s;
}

__global__ void dense_small(const float* __restrict__ in, const float* __restrict__ w,
                            const float* __restrict__ b, float* __restrict__ out,
                            int rows, int K, int N, int relu) {
  int t = blockIdx.x * 256 + threadIdx.x;
  if (t >= rows * N) return;
  int r = t / N, n = t % N;
  float s = b ? b[n] : 0.f;
  for (int k = 0; k < K; ++k) s += in[r * K + k] * w[k * N + n];
  if (relu) s = fmaxf(s, 0.f);
  out[t] = s;
}

// ---------------------------------------------------------------------------
extern "C" void kernel_launch(void* const* d_in, const int* in_sizes, int n_in,
                              void* d_out, int out_size, void* d_ws, size_t ws_size,
                              hipStream_t stream) {
  const float* x_h   = (const float*)d_in[0];
  const int*   eidx  = (const int*)d_in[1];
  const float* eattr = (const float*)d_in[2];
  const int*   nh    = (const int*)d_in[4];
  const int H = in_sizes[0] / 64;
  const int M = in_sizes[1] / 2;
  const int E = in_sizes[2] / 20;
  const int B = in_sizes[4];
  const int* fg = eidx;
  const int* rs = eidx + M;

  // Params: jax pytree flatten order (dict keys sorted).
  const float* P[64];
  for (int i = 5; i < n_in && i < 64; ++i) P[i] = (const float*)d_in[i];
  const float *a_b0 = P[5], *a_b1 = P[6], *a_init = P[7],
              *a_root0 = P[8], *a_root1 = P[9], *a_w1 = P[10];
  struct Lay { const float *b,*e1b,*e1w,*e2b,*e2w,*g,*n1b,*n1w,*n2b,*n2w,*pb,*pw; };
  Lay L1 = { P[11],P[12],P[13],P[14],P[15],P[16],P[17],P[18],P[19],P[20],P[21],P[22] };
  Lay L2 = { P[23],P[24],P[25],P[26],P[27],P[28],P[29],P[30],P[31],P[32],P[33],P[34] };
  const float *mw0 = P[35], *mb0 = P[36], *mw1 = P[37], *mb1 = P[38],
              *mw2 = P[39], *mb2 = P[40], *mw3 = P[41], *mb3 = P[42];
  const float *resb = P[43], *resg = P[44];

  // Carve workspace.
  char* wp = (char*)d_ws;
  auto carve = [&](size_t bytes) {
    char* r = wp; wp += (bytes + 255) & ~(size_t)255; return r;
  };
  float* cnt_e = (float*)carve((size_t)E * 4);
  float* cnt_h = (float*)carve((size_t)H * 4);
  int*   off   = (int*)carve((size_t)(B + 1) * 4);
  int*   seg   = (int*)carve((size_t)E * 4);
  float* eb0   = (float*)carve((size_t)E * HID * 4);
  float* eb1   = (float*)carve((size_t)E * HID * 4);
  float* hb0   = (float*)carve((size_t)H * HID * 4);
  float* hb1   = (float*)carve((size_t)H * HID * 4);
  float* hb2   = (float*)carve((size_t)H * HID * 4);
  float* gbuf  = (float*)carve((size_t)B * HID * 4);
  float* m1    = (float*)carve((size_t)B * 256 * 4);
  float* m2    = (float*)carve((size_t)B * 64 * 4);
  float* m3    = (float*)carve((size_t)B * 16 * 4);
  (void)ws_size; (void)out_size;

  auto cdiv = [](long a, long b) { return (unsigned)((a + b - 1) / b); };
  auto zero = [&](float* p, long n) {
    unsigned g = cdiv(n, 256); if (g > 4096u) g = 4096u;
    fill_zero_f<<<g, 256, 0, stream>>>(p, n);
  };
  auto gemm = [&](const float* A1, int K1, const float* A2, int K2,
                  const float* rc, const float* W, const float* bias,
                  const float* addend, float* C, int Mr, int relu) {
    gemm_bias_act<<<cdiv(Mr, 64), 256, 0, stream>>>(A1, K1, A2, K2, rc, W, bias,
                                                    addend, C, Mr, relu);
  };

  // Degrees + chain structure (fixed across layers).
  zero(cnt_e, E); zero(cnt_h, H);
  count_deg<<<cdiv(M, 256), 256, 0, stream>>>(fg, rs, cnt_h, cnt_e, M);
  build_offsets<<<1, 1, 0, stream>>>(nh, off, B);
  build_seg<<<B, 256, 0, stream>>>(off, seg);

  // --- Hypergraph layer ---
  auto run_layer = [&](const float* xin, int din, const Lay& L, float* xout) {
    zero(eb0, (long)E * HID);
    scatter_add_rows<<<cdiv((long)M * (din >> 2), 256), 256, 0, stream>>>(
        xin, fg, rs, eb0, M, din);                               // enm sums
    gemm(eb0, din, eattr, 20, cnt_e, L.e1w, L.e1b, nullptr, eb1, E, 1);
    gemm(eb1, HID, nullptr, 0, nullptr, L.e2w, L.e2b, nullptr, eb0, E, 0); // e
    zero(hb0, (long)H * HID);
    scatter_add_rows<<<cdiv((long)M * 32, 256), 256, 0, stream>>>(
        eb0, rs, fg, hb0, M, HID);                               // nm sums
    gemm(hb0, HID, nullptr, 0, cnt_h, L.n1w, L.n1b, nullptr, hb1, H, 1);
    gemm(hb1, HID, nullptr, 0, nullptr, L.n2w, L.n2b, nullptr, hb0, H, 0); // nm2
    gemm(xin, din, nullptr, 0, nullptr, L.pw, L.pb, hb0, xout, H, 1);      // relu(z)
    layernorm_rows<<<H, 128, 0, stream>>>(xout, L.g, L.b, H);
  };
  run_layer(x_h, 64, L1, hb2);
  run_layer(hb2, HID, L2, hb2);   // in-place safe: block reads rows before write

  // --- Residue features + ARMA chain conv ---
  zero(eb0, (long)E * HID);
  scatter_add_rows<<<cdiv((long)M * 32, 256), 256, 0, stream>>>(
      hb2, fg, rs, eb0, M, HID);                                 // r sums (mean folded)
  float* eb2 = hb0;   // reuse node scratch as edge scratch
  float* eb3 = hb1;
  gemm(eb0, HID, nullptr, 0, cnt_e, a_init,  nullptr, nullptr, eb1, E, 0); // r@init_w
  gemm(eb0, HID, nullptr, 0, cnt_e, a_root0, a_b0,    nullptr, eb2, E, 0); // r@root0+b0
  prop_combine<<<cdiv((long)E * 32, 256), 256, 0, stream>>>(eb1, eb2, seg, off, eb2, E);
  gemm(eb2, HID, nullptr, 0, nullptr, a_w1,  nullptr, nullptr, eb1, E, 0); // t@w1
  gemm(eb0, HID, nullptr, 0, cnt_e, a_root1, a_b1,    nullptr, eb3, E, 0); // r@root1+b1
  prop_combine<<<cdiv((long)E * 32, 256), 256, 0, stream>>>(eb1, eb3, seg, off, eb3, E);
  layernorm_rows<<<E, 128, 0, stream>>>(eb3, resg, resb, E);

  // --- Pool + MLP head ---
  pool_sum<<<B, 128, 0, stream>>>(eb3, off, gbuf);
  dense_small<<<cdiv((long)B * 256, 256), 256, 0, stream>>>(gbuf, mw0, mb0, m1, B, 128, 256, 1);
  dense_small<<<cdiv((long)B * 64, 256), 256, 0, stream>>>(m1, mw1, mb1, m2, B, 256, 64, 1);
  dense_small<<<cdiv((long)B * 16, 256), 256, 0, stream>>>(m2, mw2, mb2, m3, B, 64, 16, 1);
  dense_small<<<cdiv((long)B, 256), 256, 0, stream>>>(m3, mw3, mb3, (float*)d_out, B, 16, 1, 0);
}
